// STGNNMixer_60825326846038
// MI455X (gfx1250) — matrix operand order repaired and checked by reading, not compiled, not run on hardware
//
#include <hip/hip_runtime.h>
#include <hip/hip_bf16.h>
#include <math.h>

#define DI __device__ __forceinline__

typedef _Float16 v16h __attribute__((ext_vector_type(16)));
typedef _Float16 v8h  __attribute__((ext_vector_type(8)));
typedef float    v8f  __attribute__((ext_vector_type(8)));

union H16 { v16h v; _Float16 e[16]; v8h q[2]; };

// ---- problem constants ----
constexpr int Bc  = 32;    // batch
constexpr int Nn  = 200;   // nodes
constexpr int Ss  = 192;   // time steps
constexpr int Fin = 8;     // input features
constexpr int Hh  = 64;    // hidden
constexpr int Pp  = 48;    // horizon
constexpr int FNf = 4;     // future features
constexpr int Ktop= 10;    // top-k
constexpr int NBl = 3;     // blocks
constexpr int SH  = Ss*Hh;        // 12288
constexpr int NPad= 208;          // adjacency M pad (13*16)
constexpr int KPad= 224;          // adjacency K pad (7*32)
constexpr int FEAT= SH + Pp*Hh;   // 15360
constexpr int BN  = Bc*Nn;        // 6400

DI v8f wmma16(v16h a, v16h b, v8f c){
  return __builtin_amdgcn_wmma_f32_16x16x32_f16(false, a, false, b, (short)0, c, false, false);
}

// A fragment (16x32 f16), row-major, K-contiguous.
// lane<16: K off 0..7 & 16..23 ; lane>=16: 8..15 & 24..31.
DI v16h fragA_f16(const _Float16* base, size_t ld, int m0, int k0, int lane){
  const _Float16* p = base + (size_t)(m0 + (lane & 15))*ld + k0 + ((lane >> 4) << 3);
  H16 u; u.q[0] = *(const v8h*)p; u.q[1] = *(const v8h*)(p + 16);
  return u.v;
}
// B fragment (32x16 f16) from memory holding B^T (N rows, K contiguous).
// lane<16: K off 0..7 & 16..23 ; lane>=16: 8..15 & 24..31.
DI v16h fragB_f16(const _Float16* base, size_t ld, int n0, int k0, int lane){
  const _Float16* p = base + (size_t)(n0 + (lane & 15))*ld + k0 + ((lane >> 4) << 3);
  H16 u; u.q[0] = *(const v8h*)p; u.q[1] = *(const v8h*)(p + 16);
  return u.v;
}
DI float gelu_exact(float v){ return 0.5f*v*(1.0f + erff(v*0.70710678118654752f)); }

// ---------------- weight pre-conversion to f16 (once, L2-resident) ----------------
__global__ void k_prep_w(const float* __restrict__ tW, const float* __restrict__ gW,
                         _Float16* __restrict__ tWh, _Float16* __restrict__ gWhT){
  int idx = blockIdx.x*256 + threadIdx.x;
  if (idx < NBl*Ss*Ss){
    tWh[idx] = (_Float16)tW[idx];                 // [i][t][s] row-major, direct
  }
  int idx2 = idx - NBl*Ss*Ss;
  if (idx2 >= 0 && idx2 < NBl*Hh*Hh){
    int i = idx2 / (Hh*Hh), r = idx2 % (Hh*Hh);
    int hi = r >> 6, ho = r & 63;                  // gW: [h_in][h_out]
    gWhT[i*Hh*Hh + ho*Hh + hi] = (_Float16)gW[idx2];   // store W^T: [h_out][h_in]
  }
}

// ---------------- adjacency pipeline ----------------
__global__ void k_adj_raw(const float* __restrict__ emb, float* __restrict__ Araw){
  int idx = blockIdx.x*256 + threadIdx.x;
  if (idx >= Nn*Nn) return;
  int i = idx / Nn, j = idx % Nn;
  const float* a = emb + i*Hh; const float* b = emb + j*Hh;
  float s = 0.f;
  #pragma unroll 8
  for (int f = 0; f < Hh; ++f) s += a[f]*b[f];
  Araw[idx] = s > 0.f ? s : 0.f;
}

__global__ void k_adj_row(const float* __restrict__ Araw, const float* __restrict__ adj,
                          const float* __restrict__ alpha_p,
                          float* __restrict__ Apre, float* __restrict__ dvec){
  __shared__ float rowv[Nn];
  __shared__ float red[256];
  int i = blockIdx.x, t = threadIdx.x;
  if (t < Nn) rowv[t] = Araw[i*Nn + t];
  __syncthreads();
  float v = (t < Nn) ? rowv[t] : 0.f;
  int rank = 0;
  if (t < Nn){
    for (int l = 0; l < Nn; ++l){
      float u = rowv[l];
      rank += (u > v) || (u == v && l < t);
    }
  }
  float masked = (t < Nn && rank < Ktop) ? v : 0.f;
  red[t] = (t < Nn) ? masked : 0.f; __syncthreads();
  for (int s = 128; s > 0; s >>= 1){ if (t < s) red[t] = fmaxf(red[t], red[t+s]); __syncthreads(); }
  float mx = red[0]; __syncthreads();
  float ex = (t < Nn) ? expf(masked - mx) : 0.f;
  red[t] = ex; __syncthreads();
  for (int s = 128; s > 0; s >>= 1){ if (t < s) red[t] += red[t+s]; __syncthreads(); }
  float Z = red[0]; __syncthreads();
  float alpha = alpha_p[0];
  float Aij = 0.f;
  if (t < Nn){
    float sm = ex / Z;
    float comb = alpha*adj[i*Nn + t] + (1.f - alpha)*sm;
    Aij = (t == i) ? 1.f : comb;
    Apre[i*Nn + t] = Aij;
  }
  red[t] = (t < Nn) ? Aij : 0.f; __syncthreads();
  for (int s = 128; s > 0; s >>= 1){ if (t < s) red[t] += red[t+s]; __syncthreads(); }
  if (t == 0) dvec[i] = rsqrtf(fmaxf(red[0], 1.f));
}

__global__ void k_adj_norm(const float* __restrict__ Apre, const float* __restrict__ dvec,
                           _Float16* __restrict__ Ah){
  int idx = blockIdx.x*256 + threadIdx.x;
  if (idx >= NPad*KPad) return;
  int ip = idx / KPad, kp = idx % KPad;
  float val = 0.f;
  if (ip < Nn && kp < Nn) val = dvec[ip]*Apre[ip*Nn + kp]*dvec[kp];
  Ah[idx] = (_Float16)val;   // zero K-pad => garbage B pad multiplies by 0
}

__global__ void k_headWT(const float* __restrict__ headW, _Float16* __restrict__ wt){
  int idx = blockIdx.x*256 + threadIdx.x;
  if (idx >= Pp*FEAT) return;
  int p = idx / FEAT, j = idx % FEAT;
  wt[idx] = (_Float16)headW[(size_t)j*Pp + p];
}

// ---------------- instance-norm + projection + node embedding ----------------
__global__ void k_inproj(const float* __restrict__ x, const float* __restrict__ in_w,
                         const float* __restrict__ in_b, const float* __restrict__ projW,
                         const float* __restrict__ projb, const float* __restrict__ emb,
                         float* __restrict__ xout){
  __shared__ float xb[Ss*Fin];
  __shared__ float mu[Fin], rs[Fin];
  int bn = blockIdx.x, t = threadIdx.x;
  int n = bn % Nn;
  const float* xin = x + (size_t)bn*Ss*Fin;
  for (int idx = t; idx < Ss*Fin; idx += 256) xb[idx] = xin[idx];
  __syncthreads();
  if (t < Fin){
    float s = 0.f, sq = 0.f;
    for (int sI = 0; sI < Ss; ++sI){ float v = xb[sI*Fin + t]; s += v; sq += v*v; }
    float m = s/(float)Ss; float var = sq/(float)Ss - m*m;
    mu[t] = m; rs[t] = rsqrtf(var + 1e-5f);
  }
  __syncthreads();
  for (int idx = t; idx < Ss*Fin; idx += 256){
    int f = idx & 7;
    xb[idx] = (xb[idx]-mu[f])*rs[f]*in_w[f] + in_b[f];
  }
  __syncthreads();
  float* xo = xout + (size_t)bn*SH;
  for (int idx = t; idx < SH; idx += 256){
    int sI = idx >> 6, h = idx & 63;
    float acc = projb[h] + emb[n*Hh + h];
    const float* xr = &xb[sI*Fin];
    #pragma unroll
    for (int f = 0; f < Fin; ++f) acc += xr[f]*projW[f*Hh + h];
    xo[idx] = acc;
  }
}

// ---- fused temporal mix + GCN weight: x += gelu(W_t@x+b) ; xw = x@W_g -> xwT ----
__global__ void k_temporal_xw(const _Float16* __restrict__ Wt16,   // [192][192] f16
                              const float* __restrict__ bias,      // [192]
                              const _Float16* __restrict__ gWT,    // W_g^T [64][64] f16
                              float* __restrict__ x,
                              _Float16* __restrict__ xwT){
  __shared__ __align__(16) _Float16 Xt[Hh*200];   // X^T [h][s], stride 200 (400B=25*16)
  __shared__ __align__(16) _Float16 Xs[Ss*72];    // updated X [s][h], stride 72 (144B=9*16)
  int bn = blockIdx.x, t = threadIdx.x;
  int b = bn / Nn, n = bn % Nn;
  float* xt = x + (size_t)bn*SH;
  for (int idx = t; idx < SH; idx += 256){
    int sI = idx >> 6, h = idx & 63;
    Xt[h*200 + sI] = (_Float16)xt[idx];
  }
  __syncthreads();
  int lane = t & 31, w = t >> 5;
  // phase 1: temporal mixing, write f32 back to global + f16 into Xs
  for (int tt = w*6; tt < w*6 + 6; ++tt){      // 48 tiles = 12 M x 4 N
    int mt = tt >> 2, nt = tt & 3;
    v8f acc = {};
    #pragma unroll
    for (int k0 = 0; k0 < Ss; k0 += 32){
      v16h a = fragA_f16(Wt16, Ss, mt*16, k0, lane);   // L2-resident f16 weights
      v16h bfr = fragB_f16(Xt, 200, nt*16, k0, lane);  // LDS b128
      acc = wmma16(a, bfr, acc);
    }
    #pragma unroll
    for (int v = 0; v < 8; ++v){
      int row = mt*16 + v + ((lane >> 4) << 3);
      int col = nt*16 + (lane & 15);
      float y = acc[v] + bias[row];
      float xn = xt[row*Hh + col] + gelu_exact(y);
      xt[row*Hh + col] = xn;
      Xs[row*72 + col] = (_Float16)xn;
    }
  }
  __syncthreads();
  // phase 2: xw = x_new @ W_g, store node-contiguous for the A-mix
  _Float16* dst = xwT + (size_t)b*SH*KPad;
  for (int tt = w*6; tt < w*6 + 6; ++tt){
    int mt = tt >> 2, nt = tt & 3;
    v8f acc = {};
    #pragma unroll
    for (int k0 = 0; k0 < Hh; k0 += 32){
      v16h a = fragA_f16(Xs, 72, mt*16, k0, lane);
      v16h bfr = fragB_f16(gWT, Hh, nt*16, k0, lane);  // L2-resident f16 W^T
      acc = wmma16(a, bfr, acc);
    }
    #pragma unroll
    for (int v = 0; v < 8; ++v){
      int sI = mt*16 + v + ((lane >> 4) << 3);
      int h  = nt*16 + (lane & 15);
      dst[(size_t)(sI*Hh + h)*KPad + n] = (_Float16)acc[v];
    }
  }
}

// ---------------- g = A @ xw ; x = LN(x + g + b) fused ----------------
__global__ void k_mix_ln(const _Float16* __restrict__ Ah, const _Float16* __restrict__ xwT,
                         const float* __restrict__ gb, const float* __restrict__ lnw,
                         const float* __restrict__ lnb, float* __restrict__ x){
  __shared__ float gbuf[8*16*64];   // 32KB: per-wave 16 nodes x 64 h staging
  int t = threadIdx.x, lane = t & 31, w = t >> 5;
  int id = blockIdx.x;
  int grp = id % 24; id /= 24;
  int mt = id % 13;  int b = id / 13;
  int sI = grp*8 + w;               // each wave owns one timestep, full H
  v8f acc[4] = {};
  const _Float16* xb = xwT + (size_t)b*SH*KPad;
  for (int k0 = 0; k0 < KPad; k0 += 32){
    v16h a = fragA_f16(Ah, KPad, mt*16, k0, lane);   // A row tile reused for 4 h-tiles
    #pragma unroll
    for (int ht = 0; ht < 4; ++ht){
      v16h bfr = fragB_f16(xb, KPad, sI*Hh + ht*16, k0, lane);
      acc[ht] = wmma16(a, bfr, acc[ht]);
    }
  }
  float* gw = &gbuf[w*1024];
  #pragma unroll
  for (int ht = 0; ht < 4; ++ht){
    #pragma unroll
    for (int v = 0; v < 8; ++v){
      int m = v + ((lane >> 4) << 3);
      gw[m*64 + ht*16 + (lane & 15)] = acc[ht][v];
    }
  }
  __syncthreads();
  // LayerNorm over H=64 : 2 lanes per node-row, pair-combine with shfl_xor (wave32)
  int r = lane >> 1, hh = (lane & 1)*32;
  int node = mt*16 + r;
  if (node < Nn){
    float* xr = x + ((size_t)(b*Nn + node)*Ss + sI)*Hh;
    float s = 0.f, sq = 0.f;
    for (int j = 0; j < 32; ++j){
      int h = hh + j;
      float val = xr[h] + gw[r*64 + h] + gb[h];
      gw[r*64 + h] = val;
      s += val; sq += val*val;
    }
    s  += __shfl_xor(s, 1);
    sq += __shfl_xor(sq, 1);
    float m = s*(1.f/64.f);
    float var = sq*(1.f/64.f) - m*m;
    float rstd = rsqrtf(var + 1e-5f);
    for (int j = 0; j < 32; ++j){
      int h = hh + j;
      xr[h] = (gw[r*64 + h] - m)*rstd*lnw[h] + lnb[h];
    }
  }
}

// ---------------- feat = [x_hist | futr] as f16 rows ----------------
__global__ void k_feat(const float* __restrict__ x, const float* __restrict__ xf,
                       const float* __restrict__ fW, const float* __restrict__ fb,
                       _Float16* __restrict__ feat){
  int bn = blockIdx.x, t = threadIdx.x;
  const float* xs = x + (size_t)bn*SH;
  _Float16* fr = feat + (size_t)bn*FEAT;
  for (int idx = t; idx < SH; idx += 256) fr[idx] = (_Float16)xs[idx];
  const float* xfr = xf + (size_t)bn*Pp*FNf;
  for (int idx = t; idx < Pp*Hh; idx += 256){
    int p = idx >> 6, h = idx & 63;
    float acc = fb[h];
    #pragma unroll
    for (int f = 0; f < FNf; ++f) acc += xfr[p*FNf + f]*fW[f*Hh + h];
    fr[SH + idx] = (_Float16)acc;
  }
}

// ---------------- head: [6400 x 15360] @ [15360 x 48] ----------------
__global__ void k_head(const _Float16* __restrict__ feat, const _Float16* __restrict__ wt,
                       const float* __restrict__ hb, float* __restrict__ out){
  int t = threadIdx.x, lane = t & 31, w = t >> 5;
  int task = blockIdx.x*8 + w;        // 150 blocks * 8 waves = 400 M-tiles x 3 N-tiles
  int mt = task / 3, nt = task % 3;
  v8f acc = {};
  for (int k0 = 0; k0 < FEAT; k0 += 32){
    v16h a = fragA_f16(feat, (size_t)FEAT, mt*16, k0, lane);
    v16h b = fragB_f16(wt,   (size_t)FEAT, nt*16, k0, lane);
    acc = wmma16(a, b, acc);
  }
  #pragma unroll
  for (int v = 0; v < 8; ++v){
    int bn = mt*16 + v + ((lane >> 4) << 3);
    int p  = nt*16 + (lane & 15);
    out[(size_t)bn*Pp + p] = acc[v] + hb[p];
  }
}

extern "C" void kernel_launch(void* const* d_in, const int* in_sizes, int n_in,
                              void* d_out, int out_size, void* d_ws, size_t ws_size,
                              hipStream_t stream){
  const float* x        = (const float*)d_in[0];
  const float* x_future = (const float*)d_in[1];
  const float* adj      = (const float*)d_in[2];
  const float* emb      = (const float*)d_in[3];
  const float* alpha    = (const float*)d_in[4];
  const float* in_w     = (const float*)d_in[5];
  const float* in_b     = (const float*)d_in[6];
  const float* projW    = (const float*)d_in[7];
  const float* projb    = (const float*)d_in[8];
  const float* tempW    = (const float*)d_in[9];
  const float* tempb    = (const float*)d_in[10];
  const float* gcnW     = (const float*)d_in[11];
  const float* gcnb     = (const float*)d_in[12];
  const float* lnw      = (const float*)d_in[13];
  const float* lnb      = (const float*)d_in[14];
  const float* futrW    = (const float*)d_in[15];
  const float* futrb    = (const float*)d_in[16];
  const float* headW    = (const float*)d_in[17];
  const float* headb    = (const float*)d_in[18];
  (void)in_sizes; (void)n_in; (void)out_size; (void)ws_size;

  char* ws = (char*)d_ws;
  size_t off = 0;
  auto alloc = [&](size_t bytes) -> void* {
    void* p = ws + off;
    off += bytes;
    off = (off + 255) & ~(size_t)255;
    return p;
  };
  float*    Araw   = (float*)alloc((size_t)Nn*Nn*4);
  float*    Apre   = (float*)alloc((size_t)Nn*Nn*4);
  float*    dvec   = (float*)alloc((size_t)Nn*4);
  _Float16* Ah     = (_Float16*)alloc((size_t)NPad*KPad*2);
  _Float16* headWT = (_Float16*)alloc((size_t)Pp*FEAT*2);
  _Float16* tWh    = (_Float16*)alloc((size_t)NBl*Ss*Ss*2);
  _Float16* gWhT   = (_Float16*)alloc((size_t)NBl*Hh*Hh*2);
  float*    xcur   = (float*)alloc((size_t)BN*SH*4);
  size_t big = (size_t)Bc*SH*KPad*2;               // xw_t (node-major)
  size_t big2s = (size_t)BN*FEAT*2;                // feat (reuses same region)
  if (big2s > big) big = big2s;
  void* bigp = alloc(big);
  _Float16* xwT  = (_Float16*)bigp;
  _Float16* feat = (_Float16*)bigp;

  k_prep_w  <<<(NBl*(Ss*Ss + Hh*Hh) + 255)/256, 256, 0, stream>>>(tempW, gcnW, tWh, gWhT);
  k_adj_raw <<<(Nn*Nn + 255)/256, 256, 0, stream>>>(emb, Araw);
  k_adj_row <<<Nn, 256, 0, stream>>>(Araw, adj, alpha, Apre, dvec);
  k_adj_norm<<<(NPad*KPad + 255)/256, 256, 0, stream>>>(Apre, dvec, Ah);
  k_headWT  <<<(Pp*FEAT + 255)/256, 256, 0, stream>>>(headW, headWT);
  k_inproj  <<<BN, 256, 0, stream>>>(x, in_w, in_b, projW, projb, emb, xcur);
  for (int i = 0; i < NBl; ++i){
    k_temporal_xw<<<BN, 256, 0, stream>>>(tWh + (size_t)i*Ss*Ss, tempb + i*Ss,
                                          gWhT + (size_t)i*Hh*Hh, xcur, xwT);
    k_mix_ln     <<<Bc*13*24, 256, 0, stream>>>(Ah, xwT, gcnb + i*Hh, lnw + i*Hh,
                                                lnb + i*Hh, xcur);
  }
  k_feat<<<BN, 256, 0, stream>>>(xcur, x_future, futrW, futrb, feat);
  k_head<<<150, 256, 0, stream>>>(feat, headWT, headb, (float*)d_out);
}
